// MultiTypeEdgePooling_18769007083607
// MI455X (gfx1250) — matrix operand order repaired and checked by reading, not compiled, not run on hardware
//
#include <hip/hip_runtime.h>
#include <cstdint>
#include <cstddef>

#define T_ 3
#define E_ 500000
#define F_ 128
#define H_ 64
#define B_ 1024

typedef __attribute__((ext_vector_type(16))) _Float16 v16h;
typedef __attribute__((ext_vector_type(8)))  float    v8f;
typedef __attribute__((ext_vector_type(4)))  float    v4f;

// ---- monotone float<->uint mapping for atomic max on signed floats ----
__device__ __forceinline__ unsigned fkey(float f) {
  unsigned u = __float_as_uint(f);
  return (u & 0x80000000u) ? ~u : (u | 0x80000000u);
}
__device__ __forceinline__ float fdec(unsigned k) {
  unsigned u = (k & 0x80000000u) ? (k & 0x7fffffffu) : ~k;
  return __uint_as_float(u);
}

// ---- zero the small per-(type,graph) accumulators ----
__global__ void zero_kernel(unsigned* __restrict__ smaxu,
                            float* __restrict__ ssum,
                            unsigned* __restrict__ counts) {
  int i = blockIdx.x * blockDim.x + threadIdx.x;
  if (i < T_ * B_) { smaxu[i] = 0u; ssum[i] = 0.0f; counts[i] = 0u; }
}

// ---- Pass 1: per-edge MLP score via WMMA, + segment max via atomicMax ----
// One wave handles a 16-edge tile: A = ea[16x128] (f16), B = W1[128x64] (f16,
// 4 N-tiles x 4 K-chunks resident in registers), C accumulates f32 16x16 x4.
// ea reads are non-temporal (768 MB streamed once; keep L2 for segment state).
__global__ void __launch_bounds__(256)
score_kernel(const float* __restrict__ ea, const int* __restrict__ batch,
             const float* __restrict__ W1, const float* __restrict__ b1,
             const float* __restrict__ W2, const float* __restrict__ b2,
             float* __restrict__ score, unsigned* __restrict__ smaxu)
{
  const int t    = blockIdx.y;
  const int lane = threadIdx.x & 31;
  const int wave = threadIdx.x >> 5;
  const int nIdx = lane & 15;
  const bool hi  = lane >= 16;

  const float* __restrict__ W1t = W1 + (size_t)t * F_ * H_;
  const float  b2t = b2[t];

  // B fragments: 32x16 (KxN) f16 tile. lanes 0-15 hold K = k0+h (h=0..15),
  // lanes 16-31 hold K = k0+16+h; N = nt*16 + (lane&15).
  v16h Bf[4][4];
#pragma unroll
  for (int nt = 0; nt < 4; ++nt) {
    const int n = nt * 16 + nIdx;
#pragma unroll
    for (int kc = 0; kc < 4; ++kc) {
      const int kbase = kc * 32 + (hi ? 16 : 0);
#pragma unroll
      for (int h = 0; h < 16; ++h)
        Bf[nt][kc][h] = (_Float16)W1t[(size_t)(kbase + h) * H_ + n];
    }
  }
  float b1v[4], w2v[4];
#pragma unroll
  for (int nt = 0; nt < 4; ++nt) {
    b1v[nt] = b1[t * H_ + nt * 16 + nIdx];
    w2v[nt] = W2[t * H_ + nt * 16 + nIdx];
  }

  const int tiles = E_ / 16;               // 31250, exact
  const int gw    = blockIdx.x * (blockDim.x >> 5) + wave;
  const int gws   = gridDim.x * (blockDim.x >> 5);

  for (int tile = gw; tile < tiles; tile += gws) {
    const int ebase = tile * 16;
    // A fragment row for this lane: M = lane&15.
    const float* __restrict__ row =
        ea + ((size_t)t * E_ + (size_t)(ebase + nIdx)) * F_;
    const int coff = hi ? 16 : 0;          // lanes>=16 own K 16..31

    v16h Af[4];
#pragma unroll
    for (int kc = 0; kc < 4; ++kc) {
      float av[16];
      *(v4f*)(av + 0)  = __builtin_nontemporal_load((const v4f*)(row + kc * 32 + coff));
      *(v4f*)(av + 4)  = __builtin_nontemporal_load((const v4f*)(row + kc * 32 + coff + 4));
      *(v4f*)(av + 8)  = __builtin_nontemporal_load((const v4f*)(row + kc * 32 + coff + 8));
      *(v4f*)(av + 12) = __builtin_nontemporal_load((const v4f*)(row + kc * 32 + coff + 12));
#pragma unroll
      for (int h = 0; h < 16; ++h) Af[kc][h] = (_Float16)av[h];
    }

    v8f c[4] = {};
#pragma unroll
    for (int kc = 0; kc < 4; ++kc) {
#pragma unroll
      for (int nt = 0; nt < 4; ++nt)
        c[nt] = __builtin_amdgcn_wmma_f32_16x16x32_f16(
            false, Af[kc], false, Bf[nt][kc], (short)0, c[nt], false, false);
    }

    // Epilogue: tanh(+b1), dot with W2 in-lane, reduce across the 16 N lanes.
    // D layout: VGPR r -> M = r (lanes 0-15) or M = 8+r (lanes 16-31), N = lane&15.
    float partial[8];
#pragma unroll
    for (int r = 0; r < 8; ++r) partial[r] = 0.0f;
#pragma unroll
    for (int nt = 0; nt < 4; ++nt) {
#pragma unroll
      for (int r = 0; r < 8; ++r) {
        const float hv = tanhf(c[nt][r] + b1v[nt]);
        partial[r] += hv * w2v[nt];
      }
    }
#pragma unroll
    for (int r = 0; r < 8; ++r) {
#pragma unroll
      for (int off = 8; off >= 1; off >>= 1)
        partial[r] += __shfl_xor(partial[r], off, 32);
    }
    if (nIdx == 0) {                        // lanes 0 and 16
      const int mb = hi ? 8 : 0;
#pragma unroll
      for (int r = 0; r < 8; ++r) {
        const int e  = ebase + mb + r;
        const float s = partial[r] + b2t;
        score[(size_t)t * E_ + e] = s;
        const int bg = batch[(size_t)t * E_ + e];
        atomicMax(&smaxu[t * B_ + bg], fkey(s));
      }
    }
  }
}

// ---- Pass 2: decode segment max keys ----
__global__ void decode_max(const unsigned* __restrict__ smaxu,
                           float* __restrict__ smaxf) {
  int i = blockIdx.x * blockDim.x + threadIdx.x;
  if (i < T_ * B_) smaxf[i] = fdec(smaxu[i]);
}

// ---- Pass 3: ex = exp(score - max); segment sum + per-(t,b) edge counts ----
__global__ void exp_kernel(const int* __restrict__ batch,
                           float* __restrict__ score,       // in: score, out: ex
                           const float* __restrict__ smaxf,
                           float* __restrict__ ssum,
                           unsigned* __restrict__ counts) {
  const int t = blockIdx.y;
  const int e = blockIdx.x * blockDim.x + threadIdx.x;
  if (e >= E_) return;
  const size_t i = (size_t)t * E_ + e;
  const int bg = batch[i];
  const float ex = expf(score[i] - smaxf[t * B_ + bg]);
  score[i] = ex;
  atomicAdd(&ssum[t * B_ + bg], ex);
  atomicAdd(&counts[t * B_ + bg], 1u);
}

// ---- Pass 4: exclusive scan over the 3072 counts (single block) ----
__global__ void __launch_bounds__(1024)
scan_kernel(const unsigned* __restrict__ counts,
            unsigned* __restrict__ offsets, unsigned* __restrict__ cursor) {
  __shared__ unsigned part[1024];
  const int tid = threadIdx.x;
  const unsigned v0 = counts[3 * tid + 0];
  const unsigned v1 = counts[3 * tid + 1];
  const unsigned v2 = counts[3 * tid + 2];
  const unsigned s = v0 + v1 + v2;
  part[tid] = s;
  __syncthreads();
  for (int off = 1; off < 1024; off <<= 1) {
    unsigned x = (tid >= off) ? part[tid - off] : 0u;
    __syncthreads();
    part[tid] += x;
    __syncthreads();
  }
  const unsigned base = part[tid] - s;
  offsets[3 * tid + 0] = base;           cursor[3 * tid + 0] = base;
  offsets[3 * tid + 1] = base + v0;      cursor[3 * tid + 1] = base + v0;
  offsets[3 * tid + 2] = base + v0 + v1; cursor[3 * tid + 2] = base + v0 + v1;
}

// ---- Pass 5: counting-sort edge ids into per-(t,graph) lists ----
__global__ void build_list(const int* __restrict__ batch,
                           unsigned* __restrict__ cursor,
                           unsigned* __restrict__ list) {
  const int t = blockIdx.y;
  const int e = blockIdx.x * blockDim.x + threadIdx.x;
  if (e >= E_) return;
  const size_t i = (size_t)t * E_ + e;
  const int bg = batch[i];
  const unsigned pos = atomicAdd(&cursor[t * B_ + bg], 1u);
  list[pos] = (unsigned)i;
}

// ---- Pass 6: one block per graph; coalesced weighted gather, no atomics ----
// 2x unrolled with dual accumulators; ea reads are non-temporal (2nd stream).
__global__ void __launch_bounds__(128)
gather_kernel(const float* __restrict__ ea, const float* __restrict__ exv,
              const float* __restrict__ ssum, const unsigned* __restrict__ offsets,
              const unsigned* __restrict__ counts, const unsigned* __restrict__ list,
              float* __restrict__ out) {
  const int b = blockIdx.x;
  const int f = threadIdx.x;
  float acc0 = 0.0f, acc1 = 0.0f;
#pragma unroll
  for (int t = 0; t < T_; ++t) {
    const int idx = t * B_ + b;
    const unsigned start = offsets[idx];
    const unsigned cnt   = counts[idx];
    if (cnt == 0u) continue;
    const float inv = (1.0f / (float)T_) / ssum[idx];
    unsigned j = 0;
    for (; j + 2u <= cnt; j += 2u) {
      const unsigned id0 = list[start + j];
      const unsigned id1 = list[start + j + 1u];
      const float w0 = exv[id0] * inv;
      const float w1 = exv[id1] * inv;
      const float e0 = __builtin_nontemporal_load(ea + (size_t)id0 * F_ + f);
      const float e1 = __builtin_nontemporal_load(ea + (size_t)id1 * F_ + f);
      acc0 += w0 * e0;
      acc1 += w1 * e1;
    }
    if (j < cnt) {
      const unsigned id0 = list[start + j];
      const float w0 = exv[id0] * inv;
      acc0 += w0 * __builtin_nontemporal_load(ea + (size_t)id0 * F_ + f);
    }
  }
  out[(size_t)b * F_ + f] = acc0 + acc1;
}

extern "C" void kernel_launch(void* const* d_in, const int* in_sizes, int n_in,
                              void* d_out, int out_size, void* d_ws, size_t ws_size,
                              hipStream_t stream) {
  (void)in_sizes; (void)n_in; (void)out_size; (void)ws_size;
  const float* ea    = (const float*)d_in[0];
  const int*   batch = (const int*)d_in[1];
  const float* W1    = (const float*)d_in[2];
  const float* b1    = (const float*)d_in[3];
  const float* W2    = (const float*)d_in[4];
  const float* b2    = (const float*)d_in[5];
  float* out = (float*)d_out;

  char* ws = (char*)d_ws;
  size_t o = 0;
  auto carve = [&](size_t bytes) -> char* {
    char* p = ws + o;
    o += (bytes + 255) & ~(size_t)255;
    return p;
  };
  float*    score   = (float*)carve((size_t)T_ * E_ * sizeof(float));    // scores -> ex
  unsigned* list    = (unsigned*)carve((size_t)T_ * E_ * sizeof(unsigned));
  unsigned* smaxu   = (unsigned*)carve((size_t)T_ * B_ * sizeof(unsigned));
  float*    smaxf   = (float*)carve((size_t)T_ * B_ * sizeof(float));
  float*    ssum    = (float*)carve((size_t)T_ * B_ * sizeof(float));
  unsigned* counts  = (unsigned*)carve((size_t)T_ * B_ * sizeof(unsigned));
  unsigned* offsets = (unsigned*)carve((size_t)T_ * B_ * sizeof(unsigned));
  unsigned* cursor  = (unsigned*)carve((size_t)T_ * B_ * sizeof(unsigned));

  zero_kernel<<<(T_ * B_ + 255) / 256, 256, 0, stream>>>(smaxu, ssum, counts);

  dim3 gA(512, T_);
  score_kernel<<<gA, 256, 0, stream>>>(ea, batch, W1, b1, W2, b2, score, smaxu);

  decode_max<<<(T_ * B_ + 255) / 256, 256, 0, stream>>>(smaxu, smaxf);

  dim3 gE((E_ + 255) / 256, T_);
  exp_kernel<<<gE, 256, 0, stream>>>(batch, score, smaxf, ssum, counts);

  scan_kernel<<<1, 1024, 0, stream>>>(counts, offsets, cursor);

  build_list<<<gE, 256, 0, stream>>>(batch, cursor, list);

  gather_kernel<<<B_, 128, 0, stream>>>(ea, score, ssum, offsets, counts, list, out);
}
